// MultiheadAttentionMechanism_78546361909979
// MI455X (gfx1250) — compile-verified, matrix operand order checked by reading, not run on hardware
//
#include <hip/hip_runtime.h>
#include <hip/hip_bf16.h>

typedef __attribute__((ext_vector_type(16))) _Float16 v16h;
typedef __attribute__((ext_vector_type(4)))  _Float16 v4h;
typedef __attribute__((ext_vector_type(8)))  float    v8f;

#define LDS_STRIDE 40   // halves per LDS tile row (80B) -> bank-conflict-free frag loads

// ---- CDNA5 primitives ------------------------------------------------------

__device__ inline v8f wmma_f16(v16h a, v16h b, v8f c) {
  return __builtin_amdgcn_wmma_f32_16x16x32_f16(
      false, a, false, b, (short)0, c, false, false);
}

// Async global->LDS copy of 16 bytes (per lane). Tracked by ASYNCcnt.
__device__ inline void async_b128(void* lds_dst, const void* gsrc) {
  unsigned int l = (unsigned int)(uintptr_t)lds_dst;        // low 32 bits = LDS offset
  unsigned long long g = (unsigned long long)(uintptr_t)gsrc;
  asm volatile("global_load_async_to_lds_b128 %0, %1, off"
               :: "v"(l), "v"(g) : "memory");
}
__device__ inline void wait_async_le4() { asm volatile("s_wait_asynccnt 4" ::: "memory"); }
__device__ inline void wait_async_0()   { asm volatile("s_wait_asynccnt 0" ::: "memory"); }

// 16x32 f16 fragment from LDS tile (row-major, stride LDS_STRIDE).
// lanes 0-15: row M=lane, K {0..7,16..23}; lanes 16-31: row M=lane-16, K {8..15,24..31}.
__device__ inline v16h lds_load_frag(const _Float16* tile, int rowBase, int lane) {
  const _Float16* p = tile + (rowBase + (lane & 15)) * LDS_STRIDE + ((lane >> 4) << 3);
  v16h f;
#pragma unroll
  for (int i = 0; i < 8; ++i) f[i] = p[i];
#pragma unroll
  for (int i = 0; i < 8; ++i) f[8 + i] = p[16 + i];
  return f;
}

// ---- Kernel 0: streaming f32 -> f16 conversion -----------------------------
__global__ __launch_bounds__(256) void cvt_f32_f16(
    const float* __restrict__ in, _Float16* __restrict__ out, int n4)
{
  int i = blockIdx.x * 256 + threadIdx.x;
  if (i < n4) {
    float4 x = ((const float4*)in)[i];
    v4h h; h[0] = (_Float16)x.x; h[1] = (_Float16)x.y;
           h[2] = (_Float16)x.z; h[3] = (_Float16)x.w;
    ((v4h*)out)[i] = h;
  }
}

// ---- Kernel 1: projection  C[4096,1024] = A @ W^T + b  (f16 operands) ------
// transposeOut==0 -> [bh][1024][64] (q,k);  ==1 -> [bh][64][1024] (v^T)
__global__ __launch_bounds__(256) void proj_kernel(
    const _Float16* __restrict__ A, const _Float16* __restrict__ W,
    const float* __restrict__ bias, _Float16* __restrict__ out, int transposeOut)
{
  __shared__ _Float16 As[2][128 * LDS_STRIDE];
  __shared__ _Float16 Bs[2][128 * LDS_STRIDE];
  const int tid = threadIdx.x, wave = tid >> 5, lane = tid & 31;
  const int waveM = (wave >> 1) * 32, waveN = (wave & 1) * 64;
  const int tileM = blockIdx.x * 128, tileN = blockIdx.y * 128;
  const int ldrow = tid >> 1, ldcol = (tid & 1) * 16;

  const _Float16* aSrc = A + (long)(tileM + ldrow) * 1024 + ldcol;
  const _Float16* bSrc = W + (long)(tileN + ldrow) * 1024 + ldcol;
  _Float16* aDst0 = &As[0][ldrow * LDS_STRIDE + ldcol];
  _Float16* aDst1 = &As[1][ldrow * LDS_STRIDE + ldcol];
  _Float16* bDst0 = &Bs[0][ldrow * LDS_STRIDE + ldcol];
  _Float16* bDst1 = &Bs[1][ldrow * LDS_STRIDE + ldcol];

  auto prefetch = [&](int kt, int buf) {
    _Float16* ad = buf ? aDst1 : aDst0;
    _Float16* bd = buf ? bDst1 : bDst0;
    async_b128(ad,     aSrc + kt * 32);
    async_b128(ad + 8, aSrc + kt * 32 + 8);
    async_b128(bd,     bSrc + kt * 32);
    async_b128(bd + 8, bSrc + kt * 32 + 8);
  };

  v8f acc[2][4] = {};
  const int NK = 32;
  prefetch(0, 0);
  for (int kt = 0; kt < NK; ++kt) {
    const int cur = kt & 1;
    if (kt + 1 < NK) { prefetch(kt + 1, cur ^ 1); wait_async_le4(); }
    else             { wait_async_0(); }
    __syncthreads();
    v16h a0 = lds_load_frag(As[cur], waveM, lane);
    v16h a1 = lds_load_frag(As[cur], waveM + 16, lane);
#pragma unroll
    for (int ni = 0; ni < 4; ++ni) {
      v16h b = lds_load_frag(Bs[cur], waveN + ni * 16, lane);
      acc[0][ni] = wmma_f16(a0, b, acc[0][ni]);
      acc[1][ni] = wmma_f16(a1, b, acc[1][ni]);
    }
    __syncthreads();
  }

  const int colInLane = lane & 15, rowShift = (lane >> 4) * 8;
#pragma unroll
  for (int mi = 0; mi < 2; ++mi)
#pragma unroll
    for (int ni = 0; ni < 4; ++ni) {
      int col = tileN + waveN + ni * 16 + colInLane;
      float bv = bias[col];
      int h = col >> 6, d = col & 63;
#pragma unroll
      for (int r = 0; r < 8; ++r) {
        int row = tileM + waveM + mi * 16 + rowShift + r;
        int b = row >> 10, t = row & 1023;
        float val = acc[mi][ni][r] + bv;
        long idx = transposeOut
            ? ((((long)(b * 16 + h) * 64 + d) << 10) + t)
            : (((((long)(b * 16 + h)) << 10) + t) * 64 + d);
        out[idx] = (_Float16)val;
      }
    }
}

// ---- Kernel 2: scores  S[bh][1024][1024] = q @ k^T * 1/sqrt(64) ------------
__global__ __launch_bounds__(256) void scores_kernel(
    const _Float16* __restrict__ q, const _Float16* __restrict__ k,
    float* __restrict__ aw)
{
  __shared__ _Float16 As[2][128 * LDS_STRIDE];
  __shared__ _Float16 Bs[2][128 * LDS_STRIDE];
  const int bh = blockIdx.z;
  const int tileM = blockIdx.x * 128, tileN = blockIdx.y * 128;
  const int tid = threadIdx.x, wave = tid >> 5, lane = tid & 31;
  const int waveM = (wave >> 1) * 32, waveN = (wave & 1) * 64;
  const int ldrow = tid >> 1, ldcol = (tid & 1) * 16;

  const _Float16* aSrc = q + (long)bh * 1024 * 64 + (long)(tileM + ldrow) * 64 + ldcol;
  const _Float16* bSrc = k + (long)bh * 1024 * 64 + (long)(tileN + ldrow) * 64 + ldcol;
  _Float16* aDst0 = &As[0][ldrow * LDS_STRIDE + ldcol];
  _Float16* aDst1 = &As[1][ldrow * LDS_STRIDE + ldcol];
  _Float16* bDst0 = &Bs[0][ldrow * LDS_STRIDE + ldcol];
  _Float16* bDst1 = &Bs[1][ldrow * LDS_STRIDE + ldcol];

  auto prefetch = [&](int kt, int buf) {
    _Float16* ad = buf ? aDst1 : aDst0;
    _Float16* bd = buf ? bDst1 : bDst0;
    async_b128(ad,     aSrc + kt * 32);
    async_b128(ad + 8, aSrc + kt * 32 + 8);
    async_b128(bd,     bSrc + kt * 32);
    async_b128(bd + 8, bSrc + kt * 32 + 8);
  };

  v8f acc[2][4] = {};
  const int NK = 2;
  prefetch(0, 0);
  for (int kt = 0; kt < NK; ++kt) {
    const int cur = kt & 1;
    if (kt + 1 < NK) { prefetch(kt + 1, cur ^ 1); wait_async_le4(); }
    else             { wait_async_0(); }
    __syncthreads();
    v16h a0 = lds_load_frag(As[cur], waveM, lane);
    v16h a1 = lds_load_frag(As[cur], waveM + 16, lane);
#pragma unroll
    for (int ni = 0; ni < 4; ++ni) {
      v16h b = lds_load_frag(Bs[cur], waveN + ni * 16, lane);
      acc[0][ni] = wmma_f16(a0, b, acc[0][ni]);
      acc[1][ni] = wmma_f16(a1, b, acc[1][ni]);
    }
    __syncthreads();
  }

  float* outb = aw + (long)bh * 1024 * 1024;
  const int colInLane = lane & 15, rowShift = (lane >> 4) * 8;
#pragma unroll
  for (int mi = 0; mi < 2; ++mi)
#pragma unroll
    for (int ni = 0; ni < 4; ++ni) {
      int col = tileN + waveN + ni * 16 + colInLane;
#pragma unroll
      for (int r = 0; r < 8; ++r) {
        int row = tileM + waveM + mi * 16 + rowShift + r;
        outb[(long)row * 1024 + col] = acc[mi][ni][r] * 0.125f;  // 1/sqrt(64)
      }
    }
}

// ---- Kernel 3: row softmax in place over aw, with mask ---------------------
__global__ __launch_bounds__(256) void softmax_kernel(
    float* __restrict__ aw, const int* __restrict__ mask)
{
  const int qrow = blockIdx.x, h = blockIdx.y, b = blockIdx.z;
  float* row = aw + ((((long)(b * 16 + h)) << 10) + qrow) * 1024;
  const int* mrow = mask + ((((long)b) << 10) + qrow) * 1024;
  const int tid = threadIdx.x;

  float vals[4]; int msk[4];
  float lmax = -3.402823466e38f;
#pragma unroll
  for (int i = 0; i < 4; ++i) {
    int idx = tid + i * 256;
    msk[i] = mrow[idx];
    vals[i] = row[idx];
    if (msk[i] != 0) lmax = fmaxf(lmax, vals[i]);
  }
  __shared__ float red[256];
  red[tid] = lmax; __syncthreads();
  for (int s = 128; s > 0; s >>= 1) {
    if (tid < s) red[tid] = fmaxf(red[tid], red[tid + s]);
    __syncthreads();
  }
  float rmax = red[0];
  __syncthreads();
  float lsum = 0.f;
#pragma unroll
  for (int i = 0; i < 4; ++i) {
    float e = (msk[i] != 0) ? __expf(vals[i] - rmax) : 0.f;
    vals[i] = e; lsum += e;
  }
  red[tid] = lsum; __syncthreads();
  for (int s = 128; s > 0; s >>= 1) {
    if (tid < s) red[tid] += red[tid + s];
    __syncthreads();
  }
  float inv = 1.0f / red[0];
#pragma unroll
  for (int i = 0; i < 4; ++i) row[tid + i * 256] = vals[i] * inv;
}

// ---- Kernel 4: cv[bh][1024][64] = aw[bh] @ v  (v stored transposed) --------
__global__ __launch_bounds__(256) void av_kernel(
    const float* __restrict__ aw, const _Float16* __restrict__ vT,
    _Float16* __restrict__ cv)
{
  __shared__ _Float16 As[128 * LDS_STRIDE];
  __shared__ _Float16 Bs[64 * LDS_STRIDE];
  const int bh = blockIdx.y;
  const int tileM = blockIdx.x * 128;
  const int tid = threadIdx.x, wave = tid >> 5, lane = tid & 31;
  const int waveM = wave * 16;
  const float* awb = aw + (long)bh * 1024 * 1024;
  const _Float16* vb = vT + (long)bh * 64 * 1024;
  const int arow = tid >> 1, acol = (tid & 1) * 16;
  const int brow = tid >> 2, bcol = (tid & 3) * 8;

  v8f acc[4] = {};

  for (int k0 = 0; k0 < 1024; k0 += 32) {
    {
      const float4* s4 = (const float4*)(awb + (long)(tileM + arow) * 1024 + k0 + acol);
      _Float16* dst = As + arow * LDS_STRIDE + acol;
#pragma unroll
      for (int v = 0; v < 4; ++v) {
        float4 x = s4[v];
        dst[v * 4 + 0] = (_Float16)x.x; dst[v * 4 + 1] = (_Float16)x.y;
        dst[v * 4 + 2] = (_Float16)x.z; dst[v * 4 + 3] = (_Float16)x.w;
      }
    }
    {
      const float4* s4 = (const float4*)(vb + (long)brow * 1024 + k0 + bcol);
      *(float4*)(Bs + brow * LDS_STRIDE + bcol) = s4[0];
    }
    __syncthreads();
    v16h a = lds_load_frag(As, waveM, lane);
#pragma unroll
    for (int ni = 0; ni < 4; ++ni) {
      v16h b = lds_load_frag(Bs, ni * 16, lane);
      acc[ni] = wmma_f16(a, b, acc[ni]);
    }
    __syncthreads();
  }

  _Float16* cvb = cv + (long)bh * 1024 * 64;
  const int colInLane = lane & 15, rowShift = (lane >> 4) * 8;
#pragma unroll
  for (int ni = 0; ni < 4; ++ni) {
    int col = ni * 16 + colInLane;
#pragma unroll
    for (int r = 0; r < 8; ++r) {
      int row = tileM + waveM + rowShift + r;
      cvb[(long)row * 64 + col] = (_Float16)acc[ni][r];
    }
  }
}

// ---- Kernel 5: out[4096,1024] = cv @ Wo^T + bo  (f16 operands) -------------
__global__ __launch_bounds__(256) void outproj_kernel(
    const _Float16* __restrict__ cv, const _Float16* __restrict__ Wo,
    const float* __restrict__ bo, float* __restrict__ out)
{
  __shared__ _Float16 As[2][128 * LDS_STRIDE];
  __shared__ _Float16 Bs[2][128 * LDS_STRIDE];
  const int tid = threadIdx.x, wave = tid >> 5, lane = tid & 31;
  const int waveM = (wave >> 1) * 32, waveN = (wave & 1) * 64;
  const int tileM = blockIdx.x * 128, tileN = blockIdx.y * 128;
  const int ldrow = tid >> 1, ldcol = (tid & 1) * 16;

  const int m = tileM + ldrow;
  const int bb = m >> 10, t = m & 1023;
  const _Float16* bSrc = Wo + (long)(tileN + ldrow) * 1024 + ldcol;
  _Float16* aDst0 = &As[0][ldrow * LDS_STRIDE + ldcol];
  _Float16* aDst1 = &As[1][ldrow * LDS_STRIDE + ldcol];
  _Float16* bDst0 = &Bs[0][ldrow * LDS_STRIDE + ldcol];
  _Float16* bDst1 = &Bs[1][ldrow * LDS_STRIDE + ldcol];

  auto prefetch = [&](int kt, int buf) {
    int kg = kt * 32 + ldcol;                 // 16-aligned, stays inside one head block
    int h = kg >> 6, d = kg & 63;
    const _Float16* aSrc = cv + (((((long)(bb * 16 + h)) << 10) + t) * 64 + d);
    _Float16* ad = buf ? aDst1 : aDst0;
    _Float16* bd = buf ? bDst1 : bDst0;
    async_b128(ad,     aSrc);
    async_b128(ad + 8, aSrc + 8);
    async_b128(bd,     bSrc + kt * 32);
    async_b128(bd + 8, bSrc + kt * 32 + 8);
  };

  v8f acc[2][4] = {};
  const int NK = 32;
  prefetch(0, 0);
  for (int kt = 0; kt < NK; ++kt) {
    const int cur = kt & 1;
    if (kt + 1 < NK) { prefetch(kt + 1, cur ^ 1); wait_async_le4(); }
    else             { wait_async_0(); }
    __syncthreads();
    v16h a0 = lds_load_frag(As[cur], waveM, lane);
    v16h a1 = lds_load_frag(As[cur], waveM + 16, lane);
#pragma unroll
    for (int ni = 0; ni < 4; ++ni) {
      v16h b = lds_load_frag(Bs[cur], waveN + ni * 16, lane);
      acc[0][ni] = wmma_f16(a0, b, acc[0][ni]);
      acc[1][ni] = wmma_f16(a1, b, acc[1][ni]);
    }
    __syncthreads();
  }

  const int colInLane = lane & 15, rowShift = (lane >> 4) * 8;
#pragma unroll
  for (int mi = 0; mi < 2; ++mi)
#pragma unroll
    for (int ni = 0; ni < 4; ++ni) {
      int col = tileN + waveN + ni * 16 + colInLane;
      float bv = bo[col];
#pragma unroll
      for (int r = 0; r < 8; ++r) {
        int row = tileM + waveM + mi * 16 + rowShift + r;
        out[(long)row * 1024 + col] = acc[mi][ni][r] + bv;
      }
    }
}

// ---- Host launch -----------------------------------------------------------
extern "C" void kernel_launch(void* const* d_in, const int* in_sizes, int n_in,
                              void* d_out, int out_size, void* d_ws, size_t ws_size,
                              hipStream_t stream) {
  (void)in_sizes; (void)n_in; (void)out_size; (void)ws_size;
  const float* key_fwd   = (const float*)d_in[0];
  const float* value_fwd = (const float*)d_in[1];
  const float* query_fwd = (const float*)d_in[2];
  const int*   mask      = (const int*)  d_in[3];
  const float* Wk = (const float*)d_in[4];  const float* bk = (const float*)d_in[5];
  const float* Wv = (const float*)d_in[6];  const float* bv = (const float*)d_in[7];
  const float* Wq = (const float*)d_in[8];  const float* bq = (const float*)d_in[9];
  const float* Wo = (const float*)d_in[10]; const float* bo = (const float*)d_in[11];

  float* out = (float*)d_out;                  // cv: [4,1024,1024]
  float* aw  = out + (long)4 * 1024 * 1024;    // aw: [4,16,1024,1024]

  const long ACT = (long)4 * 1024 * 1024;      // 4,194,304 elements
  const long WEL = (long)1024 * 1024;          // 1,048,576 elements
  _Float16* p = (_Float16*)d_ws;
  _Float16* qc  = p; p += ACT;   // converted activations
  _Float16* kc  = p; p += ACT;
  _Float16* vc  = p; p += ACT;
  _Float16* Wqc = p; p += WEL;   // converted weights
  _Float16* Wkc = p; p += WEL;
  _Float16* Wvc = p; p += WEL;
  _Float16* Woc = p; p += WEL;
  _Float16* q_ws  = p; p += ACT; // per-head projections
  _Float16* k_ws  = p; p += ACT;
  _Float16* vT_ws = p; p += ACT;
  _Float16* cv_ws = p; p += ACT;

  dim3 blk(256);
  // 0) one-time f32 -> f16 conversions (streaming)
  cvt_f32_f16<<<dim3((int)(ACT / 4 / 256)), blk, 0, stream>>>(query_fwd, qc, (int)(ACT / 4));
  cvt_f32_f16<<<dim3((int)(ACT / 4 / 256)), blk, 0, stream>>>(key_fwd,   kc, (int)(ACT / 4));
  cvt_f32_f16<<<dim3((int)(ACT / 4 / 256)), blk, 0, stream>>>(value_fwd, vc, (int)(ACT / 4));
  cvt_f32_f16<<<dim3((int)(WEL / 4 / 256)), blk, 0, stream>>>(Wq, Wqc, (int)(WEL / 4));
  cvt_f32_f16<<<dim3((int)(WEL / 4 / 256)), blk, 0, stream>>>(Wk, Wkc, (int)(WEL / 4));
  cvt_f32_f16<<<dim3((int)(WEL / 4 / 256)), blk, 0, stream>>>(Wv, Wvc, (int)(WEL / 4));
  cvt_f32_f16<<<dim3((int)(WEL / 4 / 256)), blk, 0, stream>>>(Wo, Woc, (int)(WEL / 4));
  // 1) projections (async double-buffered WMMA GEMMs)
  proj_kernel<<<dim3(32, 8), blk, 0, stream>>>(qc, Wqc, bq, q_ws,  0);
  proj_kernel<<<dim3(32, 8), blk, 0, stream>>>(kc, Wkc, bk, k_ws,  0);
  proj_kernel<<<dim3(32, 8), blk, 0, stream>>>(vc, Wvc, bv, vT_ws, 1);
  // 2) raw scores
  scores_kernel<<<dim3(8, 8, 64), blk, 0, stream>>>(q_ws, k_ws, aw);
  // 3) masked softmax in place
  softmax_kernel<<<dim3(1024, 16, 4), blk, 0, stream>>>(aw, mask);
  // 4) cv = aw @ v
  av_kernel<<<dim3(8, 64), blk, 0, stream>>>(aw, vT_ws, cv_ws);
  // 5) output projection
  outproj_kernel<<<dim3(32, 8), blk, 0, stream>>>(cv_ws, Woc, bo, out);
}
